// DA_RNN_5299989643565
// MI455X (gfx1250) — matrix operand order repaired and checked by reading, not compiled, not run on hardware
//
#include <hip/hip_runtime.h>
#include <hip/hip_bf16.h>

// ---------------------------------------------------------------------------
// DA-RNN forward on MI455X (gfx1250, wave32, WMMA).
// B=1024, T=64, D=128, H=128, ATT=64, HORIZON=24, OUT=1.
// Persistent per-batch-tile kernels; fp32 cell state, fp16 WMMA operands kept
// pre-swizzled in LDS (A-fragment layout -> 2x ds_load_b128 per operand),
// weights pre-packed into WMMA B-fragment layout in workspace and deliberately
// re-streamed from L2 every step (LICM defeated) to avoid VGPR spill storms.
// ---------------------------------------------------------------------------

typedef __attribute__((ext_vector_type(16))) _Float16 v16h;
typedef __attribute__((ext_vector_type(8)))  float    v8f;

#if __has_builtin(__builtin_amdgcn_tanhf)
static __device__ __forceinline__ float fast_tanh(float x) { return __builtin_amdgcn_tanhf(x); }
#else
static __device__ __forceinline__ float fast_tanh(float x) { return tanhf(x); }
#endif

#if __has_builtin(__builtin_amdgcn_rcpf)
static __device__ __forceinline__ float fast_rcp(float x) { return __builtin_amdgcn_rcpf(x); }
#else
static __device__ __forceinline__ float fast_rcp(float x) { return 1.f / x; }
#endif

static __device__ __forceinline__ float sigf(float x) { return fast_rcp(1.f + __expf(-x)); }

static __device__ __forceinline__ v8f wmma_f16(v16h a, v16h b, v8f c) {
  // D = A(16x32 f16) * B(32x16 f16) + C(16x16 f32)
  return __builtin_amdgcn_wmma_f32_16x16x32_f16(
      /*neg_a=*/false, a, /*neg_b=*/false, b,
      /*c_mod=*/(short)0, c, /*reuse_a=*/false, /*reuse_b=*/false);
}

// A-fragment swizzle: element (m, k) of a row-major 16xK operand lives at
// frag kf = k>>5, lane = m + 16*((k&8)!=0), e = (k&7) | ((k&16)>>1).
static __device__ __forceinline__ int aswz(int m, int k) {
  const int kf = k >> 5;
  const int kk = k & 31;
  const int lane = m + ((kk & 8) ? 16 : 0);
  const int e = (kk & 7) | ((kk & 16) >> 1);
  return kf * 512 + lane * 16 + e;
}

// A-fragment from a pre-swizzled LDS mirror: 32 contiguous bytes per lane.
static __device__ __forceinline__ v16h fragA(const v16h* m, int kf, int lane) {
  return m[kf * 32 + lane];
}

// A-fragment gathered from row-major fp16 global memory (used by eproj GEMM).
static __device__ __forceinline__ v16h ldA_f16(const _Float16* s, int ld, int k0, int lane) {
  const int mm = lane & 15;
  const int kb = (lane & 16) ? 8 : 0;
  v16h a;
#pragma unroll
  for (int p = 0; p < 8; ++p) {
    const int k = k0 + ((p < 4) ? (kb + 2 * p) : (16 + kb + 2 * (p - 4)));
    a[2 * p]     = s[mm * ld + k];
    a[2 * p + 1] = s[mm * ld + k + 1];
  }
  return a;
}

// B-fragment: pre-swizzled in workspace, 512 halves per fragment,
// 32 bytes per lane -> 2x global_load_b128 from L2.
static __device__ __forceinline__ v16h fragB(const _Float16* base, int idx, int lane) {
  const v16h* p = (const v16h*)(base + (size_t)idx * 512);
  return p[lane];
}

// ---------------------------------------------------------------------------
// Pack weight W[n][k] (row-major, leading dim ld, column offset col0) into
// B-fragment layout: frag[(nt*KF+kf)][lane][e], lane -> n = nt*16+(lane&15),
// element e -> k = kf*32 + 8*(lane>=16) + (e&7) + 16*(e>>3).   (B[k][n] = W[n][k])
// ---------------------------------------------------------------------------
__global__ void pack_frags(const float* __restrict__ W, _Float16* __restrict__ dst,
                           int ntiles, int kfrags, int ld, int col0) {
  const int idx   = blockIdx.x * 256 + threadIdx.x;
  const int total = ntiles * kfrags * 512;
  if (idx >= total) return;
  const int e  = idx & 15;
  const int L  = (idx >> 4) & 31;
  const int fi = idx >> 9;
  const int kf = fi % kfrags;
  const int nt = fi / kfrags;
  const int n  = nt * 16 + (L & 15);
  const int k  = kf * 32 + ((L & 16) ? 8 : 0) + (e & 7) + ((e & 8) ? 16 : 0);
  dst[idx] = (_Float16)W[(size_t)n * ld + col0 + k];
}

// ---------------------------------------------------------------------------
// Encoder: persistent scan over T=64. One block = 16 batch rows, 256 threads.
// ---------------------------------------------------------------------------
__global__ __launch_bounds__(256) void enc_kernel(
    const float* __restrict__ X, const float* __restrict__ We_w,
    const float* __restrict__ We_b, const float* __restrict__ ve_w,
    const float* __restrict__ ve_b, const float* __restrict__ enc_bih,
    const float* __restrict__ enc_bhh,
    const _Float16* __restrict__ fWhs, const _Float16* __restrict__ fWih,
    const _Float16* __restrict__ fWhh, _Float16* __restrict__ ench) {
  __shared__ v16h     hcm[8 * 32];       // [h|c] fp16 mirror, A-frag layout (K=256)
  __shared__ v16h     xtm[4 * 32];       // x_tilde fp16, A-frag layout (K=128)
  __shared__ float    cf[16 * 128];      // fp32 cell state
  __shared__ float    base_l[16 * 64];   // attention pre-activation
  __shared__ float    attw[64];          // w_feat
  __shared__ float    attv[64];          // ve_w
  __shared__ float    rowsum[16];
  __shared__ float    region[16 * 512];  // scores (16x128) then gates (16x512)
  float* scores = region;
  float* gates  = region;
  _Float16* hm  = (_Float16*)hcm;
  _Float16* xm  = (_Float16*)xtm;

  const int tid  = threadIdx.x;
  const int lane = tid & 31;
  const int wid  = tid >> 5;
  const int b0   = blockIdx.x * 16;

  {
    const v16h z = {};
    if (tid < 256) hcm[tid] = z;
    if (tid < 128) xtm[tid] = z;
  }
  for (int i = tid; i < 16 * 128; i += 256) cf[i] = 0.f;
  if (tid < 64) { attw[tid] = We_w[tid * 257 + 256]; attv[tid] = ve_w[tid]; }
  __syncthreads();

  const float veb = ve_b[0];
  int zoff = 0;  // LICM breaker: redefined every step by empty asm

  for (int t = 0; t < 64; ++t) {
    // Re-stream weight fragments from L2 each step instead of letting LICM
    // register-cache ~30 fragments (which spills past 256 VGPRs).
    asm volatile("" : "+v"(zoff));
    const _Float16* fWhs_i = fWhs + zoff;
    const _Float16* fWih_i = fWih + zoff;
    const _Float16* fWhh_i = fWhh + zoff;

    // A: base = [h|c](16x256) @ Whs^T(256x64) + We_b   -- waves 0..3
    if (wid < 4) {
      const int nt = wid;
      v8f acc = {};
#pragma unroll
      for (int kf = 0; kf < 8; ++kf)
        acc = wmma_f16(fragA(hcm, kf, lane), fragB(fWhs_i, nt * 8 + kf, lane), acc);
      const int   n    = nt * 16 + (lane & 15);
      const int   mb   = (lane & 16) ? 8 : 0;
      const float bias = We_b[n];
#pragma unroll
      for (int r = 0; r < 8; ++r) base_l[(mb + r) * 64 + n] = acc[r] + bias;
    }
    __syncthreads();

    // B: input-attention scores over D=128 features (hw tanh energy)
    {
      const int    row   = tid >> 4;
      const int    sub   = tid & 15;
      const size_t xbase = ((size_t)(b0 + row) * 64 + t) * 128;
      if (t < 63) __builtin_prefetch(&X[xbase + 128 + sub], 0, 0);  // next step's tile
#pragma unroll
      for (int j = 0; j < 8; ++j) {
        const int   d  = sub + j * 16;
        const float xv = X[xbase + d];
        float       s  = veb;
        for (int a = 0; a < 64; ++a)
          s += attv[a] * fast_tanh(base_l[row * 64 + a] + xv * attw[a]);
        scores[row * 128 + d] = s;
      }
    }
    __syncthreads();

    // C1: softmax over features (per-row serial, 16 rows)
    if (tid < 16) {
      const int row = tid;
      float     m   = -3.4e38f;
      for (int d = 0; d < 128; ++d) m = fmaxf(m, scores[row * 128 + d]);
      float s = 0.f;
      for (int d = 0; d < 128; ++d) {
        float e = __expf(scores[row * 128 + d] - m);
        scores[row * 128 + d] = e;
        s += e;
      }
      rowsum[row] = fast_rcp(s);
    }
    __syncthreads();

    // C2: x_tilde = alpha * x, stored directly in A-frag swizzle (fp16)
    {
      const int    row   = tid >> 4;
      const int    sub   = tid & 15;
      const size_t xbase = ((size_t)(b0 + row) * 64 + t) * 128;
      const float  inv   = rowsum[row];
#pragma unroll
      for (int j = 0; j < 8; ++j) {
        const int d = sub + j * 16;
        xm[aswz(row, d)] = (_Float16)(scores[row * 128 + d] * inv * X[xbase + d]);
      }
    }
    __syncthreads();

    // D: gates(16x512) = x_tilde @ Wih^T + h @ Whh^T + bih + bhh -- 8 waves x 4 N-tiles
#pragma unroll
    for (int j = 0; j < 4; ++j) {
      const int nt  = wid * 4 + j;
      v8f       acc = {};
#pragma unroll
      for (int kf = 0; kf < 4; ++kf)
        acc = wmma_f16(fragA(xtm, kf, lane), fragB(fWih_i, nt * 4 + kf, lane), acc);
#pragma unroll
      for (int kf = 0; kf < 4; ++kf)
        acc = wmma_f16(fragA(hcm, kf, lane), fragB(fWhh_i, nt * 4 + kf, lane), acc);
      const int   n    = nt * 16 + (lane & 15);
      const int   mb   = (lane & 16) ? 8 : 0;
      const float bias = enc_bih[n] + enc_bhh[n];
#pragma unroll
      for (int r = 0; r < 8; ++r) gates[(mb + r) * 512 + n] = acc[r] + bias;
    }
    __syncthreads();

    // E: LSTM pointwise; update fp32 c, fp16 swizzled [h|c] mirror, global h2
#pragma unroll
    for (int k = 0; k < 8; ++k) {
      const int   task = tid + k * 256;
      const int   row  = task >> 7;
      const int   h    = task & 127;
      const float gi   = gates[row * 512 + h];
      const float gf   = gates[row * 512 + 128 + h];
      const float gg   = gates[row * 512 + 256 + h];
      const float go   = gates[row * 512 + 384 + h];
      const float cold = cf[row * 128 + h];
      const float c2   = sigf(gf) * cold + sigf(gi) * fast_tanh(gg);
      const float h2   = sigf(go) * fast_tanh(c2);
      cf[row * 128 + h]      = c2;
      hm[aswz(row, h)]       = (_Float16)h2;
      hm[aswz(row, 128 + h)] = (_Float16)c2;
      ench[((size_t)(b0 + row) * 64 + t) * 128 + h] = (_Float16)h2;
    }
    __syncthreads();
  }
}

// ---------------------------------------------------------------------------
// enc_proj = enc_hiddens(65536x128 f16) @ Wd_w[:, :H]^T (128x64) -> fp32
// Pure WMMA GEMM: 8 waves/block, 1 M-tile per wave, 4 N-tiles, K=4 frags.
// ---------------------------------------------------------------------------
__global__ __launch_bounds__(256) void eproj_kernel(const _Float16* __restrict__ ench,
                                                    const _Float16* __restrict__ fWdh,
                                                    float* __restrict__ eproj) {
  const int       lane = threadIdx.x & 31;
  const int       wid  = threadIdx.x >> 5;
  const int       rt   = blockIdx.x * 8 + wid;  // row-tile of 16 rows of (B*T)
  const _Float16* arow = ench + (size_t)rt * 16 * 128;
#pragma unroll
  for (int nt = 0; nt < 4; ++nt) {
    v8f acc = {};
#pragma unroll
    for (int kf = 0; kf < 4; ++kf)
      acc = wmma_f16(ldA_f16(arow, 128, kf * 32, lane), fragB(fWdh, nt * 4 + kf, lane), acc);
    const int n  = nt * 16 + (lane & 15);
    const int mb = (lane & 16) ? 8 : 0;
#pragma unroll
    for (int r = 0; r < 8; ++r)
      eproj[((size_t)(rt * 16 + mb + r)) * 64 + n] = acc[r];
  }
}

// ---------------------------------------------------------------------------
// Decoder: persistent scan over HORIZON=24. One block = 16 batch rows.
// ---------------------------------------------------------------------------
__global__ __launch_bounds__(256) void dec_kernel(
    const float* __restrict__ y_hist, const float* __restrict__ vd_w,
    const float* __restrict__ vd_b, const float* __restrict__ Wd_b,
    const float* __restrict__ dec_Wih, const float* __restrict__ dec_bih,
    const float* __restrict__ dec_bhh, const float* __restrict__ fc_w,
    const float* __restrict__ fc_b, const _Float16* __restrict__ fW2,
    const _Float16* __restrict__ fDWhh, const _Float16* __restrict__ ench,
    const float* __restrict__ eproj, float* __restrict__ out) {
  __shared__ v16h  stm_v[8 * 32];    // [d|cc] fp16 mirror, A-frag layout (K=256)
  __shared__ float ccf[16 * 128];    // fp32 cell state
  __shared__ float ctx[16 * 128];    // attention context
  __shared__ float yh[16 * 64];      // y_hist tile
  __shared__ float fcw[320];
  __shared__ float attv[64];         // vd_w
  __shared__ float yprev[16];
  __shared__ float rowsum[16];
  __shared__ float region[16 * 512]; // gates; dcv=[0:1024], sc=[1024:2048] overlap
  float* dcv    = region;
  float* sc     = region + 1024;
  float* gates  = region;
  _Float16* stm = (_Float16*)stm_v;

  const int tid  = threadIdx.x;
  const int lane = tid & 31;
  const int wid  = tid >> 5;
  const int b0   = blockIdx.x * 16;

  {
    const v16h z = {};
    if (tid < 256) stm_v[tid] = z;
  }
  for (int i = tid; i < 16 * 128; i += 256) ccf[i] = 0.f;
  for (int i = tid; i < 16 * 64; i += 256)
    yh[i] = y_hist[(size_t)(b0 + (i >> 6)) * 64 + (i & 63)];
  for (int i = tid; i < 320; i += 256) fcw[i] = fc_w[i];
  if (tid < 64) attv[tid] = vd_w[tid];
  if (tid < 16) yprev[tid] = y_hist[(size_t)(b0 + tid) * 64 + 63];
  __syncthreads();

  const float vdb = vd_b[0];
  const float fcb = fc_b[0];
  int zoff = 0;  // LICM breaker

  for (int step = 0; step < 24; ++step) {
    asm volatile("" : "+v"(zoff));
    const _Float16* fW2_i   = fW2 + zoff;
    const _Float16* fDWhh_i = fDWhh + zoff;

    // 1: dcv = [d|cc] @ W2^T + Wd_b   (W2 = Wd_w[:,128:384])
    if (wid < 4) {
      const int nt  = wid;
      v8f       acc = {};
#pragma unroll
      for (int kf = 0; kf < 8; ++kf)
        acc = wmma_f16(fragA(stm_v, kf, lane), fragB(fW2_i, nt * 8 + kf, lane), acc);
      const int   n    = nt * 16 + (lane & 15);
      const int   mb   = (lane & 16) ? 8 : 0;
      const float bias = Wd_b[n];
#pragma unroll
      for (int r = 0; r < 8; ++r) dcv[(mb + r) * 64 + n] = acc[r] + bias;
    }
    __syncthreads();

    // 2: temporal scores over T=64 (enc_proj streamed from L2, hw tanh)
#pragma unroll
    for (int k2 = 0; k2 < 4; ++k2) {
      const int    task = tid + k2 * 256;
      const int    row  = task >> 6;
      const int    tt   = task & 63;
      const float* ep   = eproj + ((size_t)(b0 + row) * 64 + tt) * 64;
      if (k2 < 3) __builtin_prefetch(ep + 256 * 64, 0, 0);  // next k2's row
      float s = vdb;
      for (int a = 0; a < 64; ++a)
        s += attv[a] * fast_tanh(ep[a] + dcv[row * 64 + a]);
      sc[row * 64 + tt] = s;
    }
    __syncthreads();

    // 3: softmax over time
    if (tid < 16) {
      const int row = tid;
      float     m   = -3.4e38f;
      for (int j = 0; j < 64; ++j) m = fmaxf(m, sc[row * 64 + j]);
      float s = 0.f;
      for (int j = 0; j < 64; ++j) {
        float e = __expf(sc[row * 64 + j] - m);
        sc[row * 64 + j] = e;
        s += e;
      }
      rowsum[row] = fast_rcp(s);
    }
    __syncthreads();

    // 4: context = beta @ enc_hiddens (fp16 from L2)
#pragma unroll
    for (int k2 = 0; k2 < 8; ++k2) {
      const int       task = tid + k2 * 256;
      const int       row  = task >> 7;
      const int       h    = task & 127;
      const _Float16* eh   = ench + (size_t)(b0 + row) * 64 * 128 + h;
      float           acc  = 0.f;
      for (int tt = 0; tt < 64; ++tt)
        acc += sc[row * 64 + tt] * (float)eh[tt * 128];
      ctx[row * 128 + h] = acc * rowsum[row];
    }
    __syncthreads();

    // 5: gates = d @ decWhh^T + y_prev*dec_Wih + bih + bhh
#pragma unroll
    for (int j = 0; j < 4; ++j) {
      const int nt  = wid * 4 + j;
      v8f       acc = {};
#pragma unroll
      for (int kf = 0; kf < 4; ++kf)
        acc = wmma_f16(fragA(stm_v, kf, lane), fragB(fDWhh_i, nt * 4 + kf, lane), acc);
      const int   n    = nt * 16 + (lane & 15);
      const int   mb   = (lane & 16) ? 8 : 0;
      const float bias = dec_bih[n] + dec_bhh[n];
      const float wih  = dec_Wih[n];
#pragma unroll
      for (int r = 0; r < 8; ++r)
        gates[(mb + r) * 512 + n] = acc[r] + bias + yprev[mb + r] * wih;
    }
    __syncthreads();

    // 6: LSTM pointwise; update fp32 cc + fp16 swizzled [d|cc] mirror
#pragma unroll
    for (int k2 = 0; k2 < 8; ++k2) {
      const int   task = tid + k2 * 256;
      const int   row  = task >> 7;
      const int   h    = task & 127;
      const float gi   = gates[row * 512 + h];
      const float gf   = gates[row * 512 + 128 + h];
      const float gg   = gates[row * 512 + 256 + h];
      const float go   = gates[row * 512 + 384 + h];
      const float cold = ccf[row * 128 + h];
      const float c2   = sigf(gf) * cold + sigf(gi) * fast_tanh(gg);
      const float h2   = sigf(go) * fast_tanh(c2);
      ccf[row * 128 + h]      = c2;
      stm[aswz(row, h)]       = (_Float16)h2;
      stm[aswz(row, 128 + h)] = (_Float16)c2;
    }
    __syncthreads();

    // 7: fc head; out feeds back as next y_prev
    if (tid < 16) {
      const int row = tid;
      float     s   = fcb;
      for (int j = 0; j < 128; ++j) s += fcw[j] * (float)stm[aswz(row, j)];
      for (int j = 0; j < 128; ++j) s += fcw[128 + j] * ctx[row * 128 + j];
      for (int j = 0; j < 64; ++j)  s += fcw[256 + j] * yh[row * 64 + j];
      out[(size_t)(b0 + row) * 24 + step] = s;
      yprev[row] = s;
    }
    __syncthreads();
  }
}

// ---------------------------------------------------------------------------
extern "C" void kernel_launch(void* const* d_in, const int* in_sizes, int n_in,
                              void* d_out, int out_size, void* d_ws, size_t ws_size,
                              hipStream_t stream) {
  (void)in_sizes; (void)n_in; (void)out_size; (void)ws_size;
  const float* X       = (const float*)d_in[0];
  const float* y_hist  = (const float*)d_in[1];
  const float* We_w    = (const float*)d_in[2];
  const float* We_b    = (const float*)d_in[3];
  const float* ve_w    = (const float*)d_in[4];
  const float* ve_b    = (const float*)d_in[5];
  const float* enc_Wih = (const float*)d_in[6];
  const float* enc_Whh = (const float*)d_in[7];
  const float* enc_bih = (const float*)d_in[8];
  const float* enc_bhh = (const float*)d_in[9];
  const float* dec_Wih = (const float*)d_in[10];
  const float* dec_Whh = (const float*)d_in[11];
  const float* dec_bih = (const float*)d_in[12];
  const float* dec_bhh = (const float*)d_in[13];
  const float* Wd_w    = (const float*)d_in[14];
  const float* Wd_b    = (const float*)d_in[15];
  const float* vd_w    = (const float*)d_in[16];
  const float* vd_b    = (const float*)d_in[17];
  const float* fc_w    = (const float*)d_in[18];
  const float* fc_b    = (const float*)d_in[19];

  char*      ws    = (char*)d_ws;
  _Float16*  fWih  = (_Float16*)(ws + 0);         // 32x4 frags = 128 KB
  _Float16*  fWhh  = (_Float16*)(ws + 131072);    // 128 KB
  _Float16*  fDWhh = (_Float16*)(ws + 262144);    // 128 KB
  _Float16*  fWhs  = (_Float16*)(ws + 393216);    // 4x8 frags = 32 KB
  _Float16*  fW2   = (_Float16*)(ws + 425984);    // 32 KB
  _Float16*  fWdh  = (_Float16*)(ws + 458752);    // 4x4 frags = 16 KB
  _Float16*  ench  = (_Float16*)(ws + 475136);    // 1024*64*128 f16 = 16 MB
  float*     eproj = (float*)(ws + 17252352);     // 1024*64*64 f32 = 16 MB

  // Pack weights into WMMA B-fragment layout (fp16).
  pack_frags<<<256, 256, 0, stream>>>(enc_Wih, fWih, 32, 4, 128, 0);
  pack_frags<<<256, 256, 0, stream>>>(enc_Whh, fWhh, 32, 4, 128, 0);
  pack_frags<<<256, 256, 0, stream>>>(dec_Whh, fDWhh, 32, 4, 128, 0);
  pack_frags<<<64, 256, 0, stream>>>(We_w, fWhs, 4, 8, 257, 0);    // We_w[:, :256]
  pack_frags<<<64, 256, 0, stream>>>(Wd_w, fW2, 4, 8, 384, 128);   // Wd_w[:, 128:384]
  pack_frags<<<32, 256, 0, stream>>>(Wd_w, fWdh, 4, 4, 384, 0);    // Wd_w[:, :128]

  enc_kernel<<<64, 256, 0, stream>>>(X, We_w, We_b, ve_w, ve_b, enc_bih, enc_bhh,
                                     fWhs, fWih, fWhh, ench);
  eproj_kernel<<<512, 256, 0, stream>>>(ench, fWdh, eproj);
  dec_kernel<<<64, 256, 0, stream>>>(y_hist, vd_w, vd_b, Wd_b, dec_Wih, dec_bih,
                                     dec_bhh, fc_w, fc_b, fW2, fDWhh, ench, eproj,
                                     (float*)d_out);
}
